// DecoderSACA_60404420051403
// MI455X (gfx1250) — compile-verified, matrix operand order checked by reading, not run on hardware
//
#include <hip/hip_runtime.h>
#include <hip/hip_bf16.h>
#include <math.h>

// Problem constants (fixed by the reference setup)
#define BB     64
#define SS     2048
#define HH     1024
#define NHEAD  16
#define HDIM   64
#define VV     50257
#define MAXLEN 1024
#define STEPK  512
#define GATES  4096
#define RC     16      // context rows per LDS chunk
#define NSPLIT 4       // split-K factor for attention
#define GWAVES 8       // waves per block in WMMA GEMM

typedef __attribute__((ext_vector_type(16))) __bf16 v16bf;
typedef __attribute__((ext_vector_type(8)))  __bf16 v8bf;
typedef __attribute__((ext_vector_type(16))) float  v16f;
typedef __attribute__((ext_vector_type(8)))  float  v8f;

// Convert one 16-element fragment of a streamed f32 operand (ISA A/B layout:
// halves 0..7 = p[0..7], halves 8..15 = p[16..23]) using native converts.
__device__ __forceinline__ v16bf cvt_frag(const float* __restrict__ p) {
  const float4* p4 = (const float4*)p;
  float4 x0 = p4[0], x1 = p4[1], x2 = p4[4], x3 = p4[5];
  v16f t = { x0.x, x0.y, x0.z, x0.w,  x1.x, x1.y, x1.z, x1.w,
             x2.x, x2.y, x2.z, x2.w,  x3.x, x3.y, x3.z, x3.w };
  return __builtin_convertvector(t, v16bf);
}

// Load a 16-element fragment from a pre-converted bf16 operand (no ALU).
__device__ __forceinline__ v16bf load_frag_bf(const __bf16* __restrict__ p) {
  v8bf lo = *(const v8bf*)(p);        // halves 0..7
  v8bf hi = *(const v8bf*)(p + 16);   // halves 8..15
  return __builtin_shufflevector(lo, hi, 0, 1, 2, 3, 4, 5, 6, 7,
                                         8, 9, 10, 11, 12, 13, 14, 15);
}

// ---------------------------------------------------------------- embedding
__global__ void k_embed(const int* __restrict__ tok,
                        const float* __restrict__ table,
                        float* __restrict__ emb, float* __restrict__ xcat) {
  int b = blockIdx.x;
  long long t = tok[b];
  const float* src = table + t * HH;
  for (int i = threadIdx.x; i < HH; i += blockDim.x) {
    float v = src[i];
    emb[b * HH + i]     = v;
    xcat[b * 2*HH + i]  = v;     // first half of LSTM input concat
  }
}

// ---------------------------------------------------- f32 -> bf16 (RNE) copy
__global__ void k_cvt_bf16(const float* __restrict__ in,
                           __bf16* __restrict__ out, int n) {
  int i = blockIdx.x * blockDim.x + threadIdx.x;
  if (i < n) out[i] = (__bf16)in[i];
}

// --------------------------------------------------- q~ = (q@Wq.T+bq)_h @ Wk_h
// scale (1/sqrt(HD)) folded into q~; K-bias dropped (softmax-invariant const)
__global__ void k_qproj(const float* __restrict__ query,
                        const float* __restrict__ Wq, const float* __restrict__ bq,
                        const float* __restrict__ Wk,
                        float* __restrict__ qt, float scale) {
  int b = blockIdx.x >> 4;
  int h = blockIdx.x & 15;
  __shared__ float Q[HDIM];
  int tid = threadIdx.x;
  if (tid < HDIM) {
    int row = h * HDIM + tid;
    float s = bq[row];
    const float* w = Wq + (long long)row * HH;
    const float* q = query + (long long)b * HH;
    for (int k = 0; k < HH; ++k) s += q[k] * w[k];
    Q[tid] = s;
  }
  __syncthreads();
  for (int d = tid; d < HH; d += blockDim.x) {
    float s = 0.f;
    #pragma unroll 8
    for (int j = 0; j < HDIM; ++j)
      s += Q[j] * Wk[(long long)(h * HDIM + j) * HH + d];
    qt[((long long)b * NHEAD + h) * HH + d] = s * scale;
  }
}

// ------------------------------------- streaming online-softmax, split-K part
// 256 threads: wave w owns heads {2w,2w+1}; 16 lanes per head, 64 dims/lane.
__global__ void __launch_bounds__(256)
k_attn_partial(const float* __restrict__ qt,
               const float* __restrict__ ctx, long long ctxBatchStride,
               int Lk, int LS,
               float* __restrict__ pacc, float* __restrict__ pm,
               float* __restrict__ pl) {
  int b    = blockIdx.x / NSPLIT;
  int sp   = blockIdx.x % NSPLIT;
  int tid  = threadIdx.x;
  int wave = tid >> 5, lane = tid & 31;
  int hh   = lane >> 4;                 // which head of the wave's pair
  int h    = wave * 2 + hh;
  int t16  = lane & 15;
  int rot  = (4 * t16 + 2 * hh) & 63;   // LDS bank stagger
  int dbase = t16 * 64;

  __shared__ float4 xs4[RC * HH / 4];   // 64 KB chunk
  float* xs = (float*)xs4;

  float qtreg[64], accr[64];
  const float* qrow = qt + ((long long)b * NHEAD + h) * HH;
  #pragma unroll
  for (int j = 0; j < 64; ++j) {
    qtreg[j] = qrow[dbase + ((j + rot) & 63)];
    accr[j]  = 0.f;
  }
  float m = -INFINITY, l = 0.f;

  int row0 = sp * LS, rowEnd = row0 + LS;
  if (rowEnd > Lk) rowEnd = Lk;
  const float* cb = ctx + (long long)b * ctxBatchStride;

  for (int r0 = row0; r0 < rowEnd; r0 += RC) {
    const float4* src4 = (const float4*)(cb + (long long)r0 * HH);
    for (int i = tid; i < RC * HH / 4; i += 256) xs4[i] = src4[i];
    if (r0 + RC < rowEnd)
      __builtin_prefetch(cb + (long long)(r0 + RC) * HH + tid * 64, 0, 0);
    __syncthreads();

    float sc[RC];
    #pragma unroll
    for (int r = 0; r < RC; ++r) {
      const float* xr = xs + r * HH + dbase;
      float p = 0.f;
      #pragma unroll
      for (int j = 0; j < 64; ++j) p += qtreg[j] * xr[(j + rot) & 63];
      p += __shfl_xor(p, 1, 16);
      p += __shfl_xor(p, 2, 16);
      p += __shfl_xor(p, 4, 16);
      p += __shfl_xor(p, 8, 16);
      sc[r] = p;
    }
    float mNew = m;
    #pragma unroll
    for (int r = 0; r < RC; ++r) mNew = fmaxf(mNew, sc[r]);
    float sOld = __expf(m - mNew);
    m = mNew; l *= sOld;
    #pragma unroll
    for (int j = 0; j < 64; ++j) accr[j] *= sOld;
    #pragma unroll
    for (int r = 0; r < RC; ++r) {
      float w = __expf(sc[r] - m);
      l += w;
      const float* xr = xs + r * HH + dbase;
      #pragma unroll
      for (int j = 0; j < 64; ++j) accr[j] += w * xr[(j + rot) & 63];
    }
    __syncthreads();
  }

  long long pb = ((long long)(b * NSPLIT + sp) * NHEAD + h) * HH;
  #pragma unroll
  for (int j = 0; j < 64; ++j) pacc[pb + dbase + ((j + rot) & 63)] = accr[j];
  if (t16 == 0) {
    pm[(b * NSPLIT + sp) * NHEAD + h] = m;
    pl[(b * NSPLIT + sp) * NHEAD + h] = l;
  }
}

// ------------------------- merge splits, normalize, apply V-projection + bias
__global__ void __launch_bounds__(256)
k_attn_combine(const float* __restrict__ pacc, const float* __restrict__ pm,
               const float* __restrict__ pl,
               const float* __restrict__ Wv, const float* __restrict__ bv,
               float* __restrict__ out, long long outStride, long long outOff) {
  int b = blockIdx.x, tid = threadIdx.x;
  int wave = tid >> 5, lane = tid & 31;
  int hh = lane >> 4, h = wave * 2 + hh, t16 = lane & 15;
  int rot = (4 * t16 + 2 * hh) & 63;
  int dbase = t16 * 64;
  __shared__ float accn[NHEAD * HH];    // 64 KB

  float M = -INFINITY;
  for (int s = 0; s < NSPLIT; ++s)
    M = fmaxf(M, pm[(b * NSPLIT + s) * NHEAD + h]);
  float L = 0.f;
  for (int s = 0; s < NSPLIT; ++s)
    L += __expf(pm[(b * NSPLIT + s) * NHEAD + h] - M) *
         pl[(b * NSPLIT + s) * NHEAD + h];
  float invL = 1.f / L;

  float acc[64];
  #pragma unroll
  for (int j = 0; j < 64; ++j) acc[j] = 0.f;
  for (int s = 0; s < NSPLIT; ++s) {
    float w = __expf(pm[(b * NSPLIT + s) * NHEAD + h] - M);
    const float* pa = pacc + ((long long)(b * NSPLIT + s) * NHEAD + h) * HH + dbase;
    #pragma unroll
    for (int j = 0; j < 64; ++j) acc[j] += w * pa[(j + rot) & 63];
  }
  #pragma unroll
  for (int j = 0; j < 64; ++j)
    accn[h * HH + dbase + ((j + rot) & 63)] = acc[j] * invL;
  __syncthreads();

  for (int n = tid; n < HH; n += 256) {
    const float* wrow = Wv + (long long)n * HH;
    const float* a = accn + (n >> 6) * HH;
    float sum = 0.f;
    for (int dd = 0; dd < HH; ++dd) {
      int d = (dd + tid) & (HH - 1);     // lane-staggered, conflict-free
      sum += wrow[d] * a[d];
    }
    out[(long long)b * outStride + outOff + n] = sum + bv[n];
  }
}

// --------------- WMMA bf16 GEMM: C(64,N) = A1·W1^T [+ A2·W2^T] + biases
// A pre-converted bf16 (row-major 64xK); W row-major f32 (N,K), converted
// in-loop with native packed converts. One wave owns a 64x16 output strip so
// every W element is loaded exactly once.
__global__ void __launch_bounds__(256)
k_gemm_bf16(const __bf16* __restrict__ A1, int K1, const float* __restrict__ W1,
            const __bf16* __restrict__ A2, int K2, const float* __restrict__ W2,
            const float* __restrict__ bias1, const float* __restrict__ bias2,
            float* __restrict__ C, int N, int ldC) {
  int wave = threadIdx.x >> 5;
  int lane = threadIdx.x & 31;
  int nt = blockIdx.x * GWAVES + wave;
  int n0 = nt * 16;
  if (n0 >= N) return;                    // wave-uniform: EXEC stays all-ones
  int ncol = n0 + (lane & 15);
  int ncl  = (ncol < N) ? ncol : (N - 1); // clamp loads in the tail tile
  int khi  = (lane < 16) ? 0 : 8;         // K sub-offset per ISA A/B layout
  int mrow = lane & 15;

  v8f acc[4] = {};

  const float* wbase1 = W1 + (long long)ncl * K1;
  for (int kb = 0; kb < K1; kb += 32) {
    v16bf bfrag = cvt_frag(wbase1 + kb + khi);
    __builtin_prefetch(wbase1 + kb + 32 + khi, 0, 0);
    #pragma unroll
    for (int mt = 0; mt < 4; ++mt) {
      v16bf afrag = load_frag_bf(A1 + (long long)(mt * 16 + mrow) * K1 + kb + khi);
      acc[mt] = __builtin_amdgcn_wmma_f32_16x16x32_bf16(
          false, afrag, false, bfrag, (short)0, acc[mt], false, false);
    }
  }
  if (A2) {
    const float* wbase2 = W2 + (long long)ncl * K2;
    for (int kb = 0; kb < K2; kb += 32) {
      v16bf bfrag = cvt_frag(wbase2 + kb + khi);
      __builtin_prefetch(wbase2 + kb + 32 + khi, 0, 0);
      #pragma unroll
      for (int mt = 0; mt < 4; ++mt) {
        v16bf afrag = load_frag_bf(A2 + (long long)(mt * 16 + mrow) * K2 + kb + khi);
        acc[mt] = __builtin_amdgcn_wmma_f32_16x16x32_bf16(
            false, afrag, false, bfrag, (short)0, acc[mt], false, false);
      }
    }
  }
  if (ncol < N) {
    float bsum = bias1 ? bias1[ncol] : 0.f;
    if (bias2) bsum += bias2[ncol];
    int mofs = (lane < 16) ? 0 : 8;       // C/D layout: VGPR r -> M=r(+8)
    #pragma unroll
    for (int mt = 0; mt < 4; ++mt)
      #pragma unroll
      for (int r = 0; r < 8; ++r)
        C[(long long)(mt * 16 + mofs + r) * ldC + ncol] = acc[mt][r] + bsum;
  }
}

// -------------------------------------------------------- LSTM element-wise
__global__ void k_lstm_elem(const float* __restrict__ gates,
                            const float* __restrict__ cin,
                            __bf16* __restrict__ hnew_bf,
                            float* __restrict__ hout, float* __restrict__ cout) {
  int idx = blockIdx.x * blockDim.x + threadIdx.x;
  if (idx >= BB * HH) return;
  int b = idx >> 10, j = idx & (HH - 1);
  const float* g = gates + (long long)b * GATES;
  float ig = 1.f / (1.f + __expf(-g[j]));
  float fg = 1.f / (1.f + __expf(-g[HH + j]));
  float gg = tanhf(g[2 * HH + j]);
  float og = 1.f / (1.f + __expf(-g[3 * HH + j]));
  float c  = cin[idx];
  float cn = fg * c + ig * gg;
  float hn = og * tanhf(cn);
  hnew_bf[idx] = (__bf16)hn;    // FC GEMM input
  hout[idx] = hn;
  cout[idx] = cn;
}

extern "C" void kernel_launch(void* const* d_in, const int* in_sizes, int n_in,
                              void* d_out, int out_size, void* d_ws, size_t ws_size,
                              hipStream_t stream) {
  const int*   tok  = (const int*)  d_in[0];
  const float* enc  = (const float*)d_in[1];
  const float* h0   = (const float*)d_in[2];
  const float* c0   = (const float*)d_in[3];
  const float* kvc  = (const float*)d_in[4];
  /* d_in[5] = decoder_step (fixed 512 by reference setup) */
  const float* embT = (const float*)d_in[6];
  const float* sa_q_w = (const float*)d_in[7],  *sa_q_b = (const float*)d_in[8];
  const float* sa_k_w = (const float*)d_in[9];
  const float* sa_v_w = (const float*)d_in[11], *sa_v_b = (const float*)d_in[12];
  const float* ca_q_w = (const float*)d_in[13], *ca_q_b = (const float*)d_in[14];
  const float* ca_k_w = (const float*)d_in[15];
  const float* ca_v_w = (const float*)d_in[17], *ca_v_b = (const float*)d_in[18];
  const float* wih = (const float*)d_in[19], *bih = (const float*)d_in[20];
  const float* whh = (const float*)d_in[21], *bhh = (const float*)d_in[22];
  const float* fcw = (const float*)d_in[23], *fcb = (const float*)d_in[24];

  float* ws     = (float*)d_ws;
  float* emb    = ws;                          // 64*1024
  float* xcat   = emb  + BB * HH;              // 64*2048
  float* sa_out = xcat + BB * 2 * HH;          // 64*1024
  float* qt     = sa_out + BB * HH;            // 64*16*1024
  float* gates  = qt   + (long long)BB * NHEAD * HH;   // 64*4096
  float* pacc   = gates + BB * GATES;          // 64*4*16*1024
  float* pm     = pacc + (long long)BB * NSPLIT * NHEAD * HH;
  float* pl     = pm   + BB * NSPLIT * NHEAD;
  __bf16* xcat_bf = (__bf16*)(pl + BB * NSPLIT * NHEAD);   // 64*2048 bf16
  __bf16* h0_bf   = xcat_bf + BB * 2 * HH;                 // 64*1024 bf16
  __bf16* hnew_bf = h0_bf + BB * HH;                       // 64*1024 bf16

  float* logits = (float*)d_out;
  float* hout   = logits + (long long)BB * VV;
  float* cout   = hout   + (long long)BB * HH;
  const float rs = 0.125f;                     // 1/sqrt(64)

  k_embed<<<BB, 256, 0, stream>>>(tok, embT, emb, xcat);
  k_cvt_bf16<<<(BB * HH + 255) / 256, 256, 0, stream>>>(h0, h0_bf, BB * HH);

  // self-attention over kv_cache[:, :512, :]
  k_qproj<<<BB * NHEAD, 256, 0, stream>>>(emb, sa_q_w, sa_q_b, sa_k_w, qt, rs);
  k_attn_partial<<<BB * NSPLIT, 256, 0, stream>>>(
      qt, kvc, (long long)MAXLEN * HH, STEPK, STEPK / NSPLIT, pacc, pm, pl);
  k_attn_combine<<<BB, 256, 0, stream>>>(pacc, pm, pl, sa_v_w, sa_v_b,
                                         sa_out, HH, 0);

  // cross-attention over encoder_outputs
  k_qproj<<<BB * NHEAD, 256, 0, stream>>>(sa_out, ca_q_w, ca_q_b, ca_k_w, qt, rs);
  k_attn_partial<<<BB * NSPLIT, 256, 0, stream>>>(
      qt, enc, (long long)SS * HH, SS, SS / NSPLIT, pacc, pm, pl);
  k_attn_combine<<<BB, 256, 0, stream>>>(pacc, pm, pl, ca_v_w, ca_v_b,
                                         xcat, 2 * HH, HH);   // second half of concat

  // concat -> bf16 once, then WMMA gate GEMM:
  // gates = xcat(64,2048)@wih^T + h0(64,1024)@whh^T + bih + bhh
  k_cvt_bf16<<<(BB * 2 * HH + 255) / 256, 256, 0, stream>>>(xcat, xcat_bf, BB * 2 * HH);
  k_gemm_bf16<<<(GATES / 16) / GWAVES, 256, 0, stream>>>(
      xcat_bf, 2 * HH, wih, h0_bf, HH, whh, bih, bhh, gates, GATES, GATES);
  k_lstm_elem<<<(BB * HH + 255) / 256, 256, 0, stream>>>(gates, c0, hnew_bf, hout, cout);

  // vocab projection: hnew(64,1024)@fcw^T + fcb  (N=50257 with tail guard)
  int fcTiles = (VV + 15) / 16;
  k_gemm_bf16<<<(fcTiles + GWAVES - 1) / GWAVES, 256, 0, stream>>>(
      hnew_bf, HH, fcw, nullptr, 0, nullptr, fcb, nullptr, logits, VV, VV);
}
